// Sampler_49770081026562
// MI455X (gfx1250) — matrix operand order repaired and checked by reading, not compiled, not run on hardware
//
#include <hip/hip_runtime.h>
#include <hip/hip_bf16.h>
#include <math.h>

#define VOC   32000
#define DK    4096
#define NROWS 128
#define NTOK  64
#define NSORT 32768
#define NT    1024
#define KC    512                  // K-chunk staged per async batch
#define NCHUNK (DK / KC)           // 8
#define NEG_INF (-__builtin_inff())

typedef __attribute__((ext_vector_type(16))) __bf16 v16bf;
typedef __attribute__((ext_vector_type(8)))  float  v8f;

// ---------------------------------------------------------------------------
// Workspace zero + token bin-count kernels
// ---------------------------------------------------------------------------
__global__ void zero_counts_kernel(unsigned* __restrict__ p, int n) {
  int i = blockIdx.x * blockDim.x + threadIdx.x;
  if (i < n) p[i] = 0u;
}

__global__ void count_tokens_kernel(const int* __restrict__ tokens,
                                    unsigned* __restrict__ counts) {
  const int row = blockIdx.x;
  const int t = threadIdx.x;           // 64 threads
  const int tok = tokens[row * NTOK + t];
  if (tok >= 0 && tok < VOC)
    atomicAdd(&counts[(size_t)row * VOC + tok], 1u);
}

// ---------------------------------------------------------------------------
// CDNA5 async HBM->LDS staging (ASYNCcnt-tracked, no VGPR round trip).
// Each call issues 8 wave-level global_load_async_to_lds_b128 per wave
// covering one 16 x KC fp32 chunk of the E tile.
// ---------------------------------------------------------------------------
__device__ __forceinline__ void stage_chunk_async(const float* __restrict__ E,
                                                  float* __restrict__ buf,
                                                  int colBase, int kBase) {
  const int tid = threadIdx.x;
#pragma unroll
  for (int it = 0; it < (16 * KC / 4) / 256; ++it) {   // 8 iterations
    const int i   = it * 256 + tid;                    // float4 index in chunk
    const int row = i >> 7;                            // / (KC/4) = /128
    const int c4  = i & 127;
    const float* g = E + (size_t)(colBase + row) * DK + kBase + c4 * 4;
    const unsigned l = (unsigned)(size_t)(buf + row * KC + c4 * 4);
    asm volatile("global_load_async_to_lds_b128 %0, %1, off"
                 :: "v"(l), "v"(g) : "memory");
  }
}

// ---------------------------------------------------------------------------
// GEMM: logits = H @ E^T + bias, fused penalty + temperature epilogue.
// One block = one 16-column tile of E. Double-buffered async staging of
// 16 x 512 fp32 chunks into LDS, consumed by v_wmma_f32_16x16x32_bf16.
// ---------------------------------------------------------------------------
__global__ __launch_bounds__(256) void gemm_logits_kernel(
    const float* __restrict__ H, const float* __restrict__ E,
    const float* __restrict__ bias,
    const float* __restrict__ pres, const float* __restrict__ freq,
    const float* __restrict__ rep,  const float* __restrict__ temp,
    const unsigned* __restrict__ counts, float* __restrict__ out)
{
  extern __shared__ char gsm[];
  float* Bs = (float*)gsm;                 // 2 x (16 x KC) fp32 = 64KB
  const int colBase = blockIdx.x * 16;
  const int tid = threadIdx.x;

  const int wave = tid >> 5;
  const int lane = tid & 31;
  const int half = lane >> 4;              // 0: lanes 0-15, 1: lanes 16-31
  const int lmod = lane & 15;
  const int rowBase = wave * 16;           // 8 waves * 16 rows = 128 rows

  const float* Arow = H + (size_t)(rowBase + lmod) * DK;

  // prime the pipeline with chunk 0
  stage_chunk_async(E, Bs, colBase, 0);

  v8f acc = {0.f, 0.f, 0.f, 0.f, 0.f, 0.f, 0.f, 0.f};
  for (int ch = 0; ch < NCHUNK; ch++) {
    float* cur = Bs + (ch & 1) * (16 * KC);
    if (ch + 1 < NCHUNK) {
      stage_chunk_async(E, Bs + ((ch + 1) & 1) * (16 * KC), colBase,
                        (ch + 1) * KC);
      // 8 ops of chunk ch+1 outstanding; async loads retire in order,
      // so waiting <=8 means chunk ch has fully landed in LDS.
      asm volatile("s_wait_asynccnt 0x8" ::: "memory");
    } else {
      asm volatile("s_wait_asynccnt 0x0" ::: "memory");
    }
    __syncthreads();                       // all waves' portions visible

    const float* BrowF = cur + lmod * KC;
    const float* Ach   = Arow + ch * KC;
#pragma unroll 2
    for (int kk = 0; kk < KC; kk += 32) {
      v16bf a, b;
      const int ka = kk + half * 8;        // A 16x32: halves K 0-7/16-23 vs 8-15/24-31
      const int kb = kk + half * 8;        // B 32x16: halves K 0-7/16-23 vs 8-15/24-31
#pragma unroll
      for (int t = 0; t < 8; t++) {
        a[t]     = (__bf16)Ach[ka + t];
        a[t + 8] = (__bf16)Ach[ka + 16 + t];
      }
#pragma unroll
      for (int t = 0; t < 8; t++) {
        b[t]     = (__bf16)BrowF[kb + t];
        b[t + 8] = (__bf16)BrowF[kb + 16 + t];
      }
      acc = __builtin_amdgcn_wmma_f32_16x16x32_bf16(false, a, false, b,
                                                    (short)0, acc, false, false);
    }
    __syncthreads();                       // done reading cur before it is re-staged
  }

  // Epilogue: bias + frequency/presence/repetition penalties + temperature.
  const int col = colBase + lmod;
  const float bv = bias[col];
#pragma unroll
  for (int r = 0; r < 8; r++) {
    const int row = rowBase + r + half * 8;   // C/D layout: VGPR r -> M=r (+8 hi half)
    float x = acc[r] + bv;
    const float cnt = (float)counts[(size_t)row * VOC + col];
    x -= freq[row] * cnt;
    const float pm = (cnt > 0.f) ? 1.f : 0.f;
    x -= pres[row] * pm;
    const float rp = rep[row];
    // branch-free repetition penalty: x>0 -> x/rp, x<0 -> x*rp (only if pm)
    const float scale = (pm > 0.f) ? ((x > 0.f) ? (1.f / rp)
                                                : ((x < 0.f) ? rp : 1.f))
                                   : 1.f;
    x *= scale;
    x /= temp[row];
    out[(size_t)row * VOC + col] = x;
  }
}

// ---------------------------------------------------------------------------
// Sampler helpers (1024 threads, 32 elements/thread, row in LDS)
// ---------------------------------------------------------------------------
__device__ __forceinline__ float scanRed(float s, float* red) {
  const int tid = threadIdx.x;
  __syncthreads();
  red[tid] = s;
  __syncthreads();
  for (int off = 1; off < NT; off <<= 1) {
    float v = red[tid];
    float add = (tid >= off) ? red[tid - off] : 0.f;
    __syncthreads();
    red[tid] = v + add;
    __syncthreads();
  }
  return red[tid];                         // inclusive over threads; red[NT-1] = total
}

__device__ __forceinline__ float sumRed(float s, float* red) {
  const int tid = threadIdx.x;
  __syncthreads();
  red[tid] = s;
  __syncthreads();
  for (int off = NT / 2; off > 0; off >>= 1) {
    if (tid < off) red[tid] += red[tid + off];
    __syncthreads();
  }
  return red[0];
}

__device__ __forceinline__ float maxRed(float s, float* red) {
  const int tid = threadIdx.x;
  __syncthreads();
  red[tid] = s;
  __syncthreads();
  for (int off = NT / 2; off > 0; off >>= 1) {
    if (tid < off) red[tid] = fmaxf(red[tid], red[tid + off]);
    __syncthreads();
  }
  return red[0];
}

// MODE 0: sort descending by value.  MODE 1: sort ascending by |value|.
template <int MODE>
__device__ void sortNet(float* vals, unsigned* idx) {
  const int tid = threadIdx.x;
  for (int k = 2; k <= NSORT; k <<= 1) {
    for (int j = k >> 1; j > 0; j >>= 1) {
      __syncthreads();
      for (int i = tid; i < NSORT; i += NT) {
        const int l = i ^ j;
        if (l > i) {
          const float a = vals[i];
          const float b = vals[l];
          const float kaf = (MODE == 0) ? -a : fabsf(a);
          const float kbf = (MODE == 0) ? -b : fabsf(b);
          const bool asc = ((i & k) == 0);
          if (asc == (kaf > kbf)) {
            vals[i] = b; vals[l] = a;
            const unsigned t = idx[i]; idx[i] = idx[l]; idx[l] = t;
          }
        }
      }
    }
  }
  __syncthreads();
}

// Stable partition: survivors (val > -inf) to front, dead to back, order kept.
__device__ void compact(float* vals, unsigned* idx, float* red) {
  const int tid = threadIdx.x;
  const int base = tid * 32;
  float lv[32]; unsigned li[32]; float lcs[32];
  float s = 0.f;
  __syncthreads();
#pragma unroll
  for (int q = 0; q < 32; q++) {
    lv[q] = vals[base + q]; li[q] = idx[base + q];
    s += (lv[q] > NEG_INF) ? 1.f : 0.f;
    lcs[q] = s;
  }
  const float inc  = scanRed(s, red);
  const float excl = inc - s;
  const float total = red[NT - 1];
#pragma unroll
  for (int q = 0; q < 32; q++) {
    const int j = base + q;
    const float rank = lcs[q] + excl;      // inclusive survivor count through j
    int pos;
    if (lv[q] > NEG_INF) pos = (int)rank - 1;
    else                 pos = (int)total + (j - (int)rank);
    vals[pos] = lv[q]; idx[pos] = li[q];
  }
  __syncthreads();
}

// ---------------------------------------------------------------------------
// Per-row sampling pipeline. P holds logits on entry, probs on exit.
// ---------------------------------------------------------------------------
__global__ __launch_bounds__(NT) void sample_kernel(
    float* __restrict__ P,
    const float* __restrict__ top_ps, const float* __restrict__ top_as,
    const float* __restrict__ tfss,   const float* __restrict__ etas,
    const float* __restrict__ epss,   const float* __restrict__ typs,
    const int* __restrict__ top_ks)
{
  extern __shared__ char smem[];
  float*    vals = (float*)smem;                                      // 128KB
  unsigned* idx  = (unsigned*)(smem + (size_t)NSORT * sizeof(float)); // 128KB
  __shared__ float    red[NT];
  __shared__ unsigned tmask[NSORT / 32];

  const int row = blockIdx.x;
  const int tid = threadIdx.x;
  const int base = tid * 32;
  float* L = P + (size_t)row * VOC;

  const float top_p = top_ps[row];
  const float top_a = top_as[row];
  const float tfs   = tfss[row];
  const float eta_c = etas[row];
  const float eps_c = epss[row];
  const float typ_p = typs[row];
  const int   top_k = top_ks[row];

  // ---- load row, pad to 32768 ----
  for (int i = tid; i < NSORT; i += NT) {
    vals[i] = (i < VOC) ? L[i] : NEG_INF;
    idx[i]  = (unsigned)i;
  }
  sortNet<0>(vals, idx);

  // ---- top-a / top-p / top-k in sorted domain ----
  {
    const float m0 = vals[0];
    float lp[32], lcs[32];
    float s = 0.f;
#pragma unroll
    for (int q = 0; q < 32; q++) {
      const float v = vals[base + q];
      const float p = (v > NEG_INF) ? __expf(v - m0) : 0.f;
      lp[q] = p; s += p; lcs[q] = s;
    }
    const float inc  = scanRed(s, red);
    const float excl = inc - s;
    const float Z    = red[NT - 1];
    const float thr  = top_a / (Z * Z);        // ps[0]^2 * top_a, ps[0] = 1/Z
#pragma unroll
    for (int q = 0; q < 32; q++) {
      const int j = base + q;
      const float p    = lp[q] / Z;
      const float csum = (lcs[q] + excl) / Z;
      bool mk = (j > 0) && ((p < thr) || (csum > top_p));
      mk = mk || (j >= top_k);
      if (mk) vals[j] = NEG_INF;
    }
  }
  compact(vals, idx, red);                     // == reference's second sort

  // ---- tail-free sampling ----
  {
    const float m1 = vals[0];
    float s = 0.f;
#pragma unroll
    for (int q = 0; q < 32; q++) {
      const float v = vals[base + q];
      s += (v > NEG_INF) ? __expf(v - m1) : 0.f;
    }
    const float Z = sumRed(s, red);
    float lcs[32];
    float s2 = 0.f;
#pragma unroll
    for (int q = 0; q < 32; q++) {
      const int i = base + q;
      float d2 = 0.f;
      if (i < VOC - 2) {
        const float v0 = vals[i], v1 = vals[i + 1], v2 = vals[i + 2];
        const float p0 = (v0 > NEG_INF) ? __expf(v0 - m1) / Z : 0.f;
        const float p1 = (v1 > NEG_INF) ? __expf(v1 - m1) / Z : 0.f;
        const float p2 = (v2 > NEG_INF) ? __expf(v2 - m1) / Z : 0.f;
        d2 = fabsf(p0 - 2.f * p1 + p2);
      }
      s2 += d2; lcs[q] = s2;
    }
    const float inc  = scanRed(s2, red);
    const float excl = inc - s2;
    const float D    = red[NT - 1];
#pragma unroll
    for (int q = 0; q < 32; q++) {
      const int i = base + q;
      if (i < VOC - 2) {
        const float cdf = (lcs[q] + excl) / D;
        if (cdf > tfs) vals[i + 1] = NEG_INF;
      }
    }
    if (tid == 0) vals[VOC - 1] = NEG_INF;     // trailing "always mask last"
  }
  compact(vals, idx, red);

  // ---- eta cutoff (suffix mask; sorted order preserved) ----
  {
    const float m2 = vals[0];
    float s = 0.f;
#pragma unroll
    for (int q = 0; q < 32; q++) {
      const float v = vals[base + q];
      s += (v > NEG_INF) ? __expf(v - m2) : 0.f;
    }
    const float Z   = sumRed(s, red);
    const float lse = __logf(Z);
    float s2 = 0.f;
#pragma unroll
    for (int q = 0; q < 32; q++) {
      const float v = vals[base + q];
      if (v > NEG_INF) { const float sh = v - m2 - lse; s2 += __expf(sh) * sh; }
    }
    const float ne  = sumRed(s2, red);
    const float eta = eta_c * 1e-4f;
    const float thr = fminf(eta, sqrtf(eta) * __expf(ne));
    const float pmax = 1.f / Z;
    const bool allmask = (pmax < thr);
#pragma unroll
    for (int q = 0; q < 32; q++) {
      const int j = base + q;
      const float v = vals[j];
      const float p = (v > NEG_INF) ? __expf(v - m2) / Z : 0.f;
      const bool mk = allmask ? (p < pmax) : (p < thr);
      if (mk) vals[j] = NEG_INF;
    }
    __syncthreads();
  }

  // ---- epsilon cutoff ----
  {
    const float m3 = vals[0];
    float s = 0.f;
#pragma unroll
    for (int q = 0; q < 32; q++) {
      const float v = vals[base + q];
      s += (v > NEG_INF) ? __expf(v - m3) : 0.f;
    }
    const float Z = sumRed(s, red);
    const float thr = eps_c * 1e-4f;
    const float pmax = 1.f / Z;
    const bool allmask = (pmax < thr);
#pragma unroll
    for (int q = 0; q < 32; q++) {
      const int j = base + q;
      const float v = vals[j];
      const float p = (v > NEG_INF) ? __expf(v - m3) / Z : 0.f;
      const bool mk = allmask ? (p < pmax) : (p < thr);
      if (mk) vals[j] = NEG_INF;
    }
    __syncthreads();
  }

  // ---- typical sampling: sort by |neg_ent - shifted| ----
  {
    const float m4 = vals[0];
    float s = 0.f;
#pragma unroll
    for (int q = 0; q < 32; q++) {
      const float v = vals[base + q];
      s += (v > NEG_INF) ? __expf(v - m4) : 0.f;
    }
    const float Z   = sumRed(s, red);
    const float lse = __logf(Z);
    float s2 = 0.f;
#pragma unroll
    for (int q = 0; q < 32; q++) {
      const float v = vals[base + q];
      if (v > NEG_INF) { const float sh = v - m4 - lse; s2 += __expf(sh) * sh; }
    }
    const float ne = sumRed(s2, red);

    float lv[32]; unsigned li[32];
#pragma unroll
    for (int q = 0; q < 32; q++) { lv[q] = vals[base + q]; li[q] = idx[base + q]; }
    __syncthreads();
#pragma unroll
    for (int q = 0; q < 32; q++) {
      const int j = base + q;
      const float v = lv[q];
      vals[j] = (v > NEG_INF) ? (ne - (v - m4 - lse)) : __builtin_inff(); // signed dev
      idx[j]  = (unsigned)j;                                             // payload = pos
    }
    tmask[tid] = 0u;
    sortNet<1>(vals, idx);                     // ascending by |dev|

    float lcs[32];
    float s3 = 0.f;
#pragma unroll
    for (int q = 0; q < 32; q++) {
      const float sg = vals[base + q];
      const float p  = __expf(ne - sg);        // prob recovered from signed dev
      s3 += p; lcs[q] = s3;
    }
    const float inc  = scanRed(s3, red);
    const float excl = inc - s3;
#pragma unroll
    for (int q = 0; q < 32; q++) {
      const int j = base + q;
      const float csum = lcs[q] + excl;
      if ((j > 0) && (csum >= typ_p)) {
        const unsigned pos = idx[j];
        atomicOr(&tmask[pos >> 5], 1u << (pos & 31));
      }
    }
    __syncthreads();
#pragma unroll
    for (int q = 0; q < 32; q++) {
      const int j = base + q;
      const bool mk = (tmask[j >> 5] >> (j & 31)) & 1u;
      vals[j] = mk ? NEG_INF : lv[q];
      idx[j]  = li[q];
    }
    __syncthreads();
  }

  // ---- final softmax, scatter to original token order ----
  {
    float mx = NEG_INF;
#pragma unroll
    for (int q = 0; q < 32; q++) mx = fmaxf(mx, vals[base + q]);
    const float M = maxRed(mx, red);
    float s = 0.f;
#pragma unroll
    for (int q = 0; q < 32; q++) {
      const float v = vals[base + q];
      s += (v > NEG_INF) ? __expf(v - M) : 0.f;
    }
    const float Z = sumRed(s, red);
#pragma unroll
    for (int q = 0; q < 32; q++) {
      const int j = base + q;
      const float v = vals[j];
      const unsigned oi = idx[j];
      if (oi < VOC) L[oi] = (v > NEG_INF) ? __expf(v - M) / Z : 0.f;
    }
  }
}

// ---------------------------------------------------------------------------
extern "C" void kernel_launch(void* const* d_in, const int* in_sizes, int n_in,
                              void* d_out, int out_size, void* d_ws, size_t ws_size,
                              hipStream_t stream) {
  const float* H      = (const float*)d_in[0];
  const float* E      = (const float*)d_in[1];
  const float* bias   = (const float*)d_in[2];
  const float* pres   = (const float*)d_in[3];
  const float* freq   = (const float*)d_in[4];
  const float* rep    = (const float*)d_in[5];
  const float* temp   = (const float*)d_in[6];
  const float* top_ps = (const float*)d_in[7];
  const float* top_as = (const float*)d_in[8];
  const float* tfss   = (const float*)d_in[9];
  const float* etas   = (const float*)d_in[10];
  const float* epss   = (const float*)d_in[11];
  const float* typs   = (const float*)d_in[12];
  const int*   toks   = (const int*)d_in[13];
  const int*   top_ks = (const int*)d_in[14];
  float* out = (float*)d_out;                 // logits scratch, then final probs
  unsigned* counts = (unsigned*)d_ws;         // 128*32000 u32 = 16.4MB

  const size_t gemm_smem   = (size_t)2 * 16 * KC * sizeof(float);              // 64KB
  const size_t sample_smem = (size_t)NSORT * (sizeof(float) + sizeof(unsigned)); // 256KB
  (void)hipFuncSetAttribute((const void*)gemm_logits_kernel,
                            hipFuncAttributeMaxDynamicSharedMemorySize, (int)gemm_smem);
  (void)hipFuncSetAttribute((const void*)sample_kernel,
                            hipFuncAttributeMaxDynamicSharedMemorySize, (int)sample_smem);

  const int ncnt = NROWS * VOC;
  zero_counts_kernel<<<(ncnt + 255) / 256, 256, 0, stream>>>(counts, ncnt);
  count_tokens_kernel<<<NROWS, NTOK, 0, stream>>>(toks, counts);
  gemm_logits_kernel<<<VOC / 16, 256, gemm_smem, stream>>>(
      H, E, bias, pres, freq, rep, temp, counts, out);
  sample_kernel<<<NROWS, NT, sample_smem, stream>>>(
      out, top_ps, top_as, tfss, etas, epss, typs, top_ks);
}
